// GroupedQueryAttention_15693810499751
// MI455X (gfx1250) — compile-verified, matrix operand order checked by reading, not run on hardware
//
#include <hip/hip_runtime.h>
#include <hip/hip_bf16.h>
#include <math.h>

#define NHEAD 9
#define NKV 3
#define HD 64
#define BB 4
#define TT 2048
#define CC 576
#define THETA 100000.0f

typedef __attribute__((ext_vector_type(16))) _Float16 v16h;
typedef __attribute__((ext_vector_type(8)))  _Float16 v8h;
typedef __attribute__((ext_vector_type(8)))  float    v8f;
typedef __attribute__((ext_vector_type(4)))  float    v4f;

// A/B fragment builder: elements 0..7 from p0 (8 contiguous f16), 8..15 from p1.
__device__ inline v16h frag_f16(const _Float16* p0, const _Float16* p1) {
  v8h lo = *(const v8h*)p0;
  v8h hi = *(const v8h*)p1;
  v16h r;
#pragma unroll
  for (int i = 0; i < 8; ++i) { r[i] = lo[i]; r[8 + i] = hi[i]; }
  return r;
}
__device__ inline v8f wmma_f16(v16h a, v16h b, v8f c) {
  return __builtin_amdgcn_wmma_f32_16x16x32_f16(false, a, false, b, (short)0, c,
                                                false, false);
}
__device__ inline float readlane_f(float v, int l) {
  return __int_as_float(__builtin_amdgcn_readlane(__float_as_int(v), l));
}

// ---------------------------------------------------------------------------
// Kernel 0: bulk f32 -> f16 conversion (8 elements / thread, b128 in/out).
// ---------------------------------------------------------------------------
__global__ __launch_bounds__(256) void cvt_f32_to_f16(
    const float* __restrict__ src, _Float16* __restrict__ dst, int nvec8)
{
  int i = blockIdx.x * 256 + threadIdx.x;
  if (i >= nvec8) return;
  v4f a = ((const v4f*)src)[2 * i];
  v4f b = ((const v4f*)src)[2 * i + 1];
  v8h d;
#pragma unroll
  for (int j = 0; j < 4; ++j) { d[j] = (_Float16)a[j]; d[4 + j] = (_Float16)b[j]; }
  ((v8h*)dst)[i] = d;
}

// ---------------------------------------------------------------------------
// Kernel 1: fused QKV projection + RoPE, all-f16 operands.
// One wave per (16-row tile, head group): g 0..8 q, 9..11 k, 12..14 v.
// q,k: f16 [B,H,T,D] (LDS-transposed, packed b128 stores).
// v:   f16 [B,KV,D,T] transposed (packed b128 stores directly).
// ---------------------------------------------------------------------------
__global__ __launch_bounds__(32) void qkv_rope_kernel(
    const _Float16* __restrict__ xh,  const _Float16* __restrict__ wqh,
    const _Float16* __restrict__ wkh, const _Float16* __restrict__ wvh,
    _Float16* __restrict__ qb, _Float16* __restrict__ kbuf,
    _Float16* __restrict__ vtb)
{
  __shared__ __align__(16) _Float16 tls[16 * 72];   // 16 rows, stride 72 f16

  const int NG = NHEAD + 2 * NKV;           // 15 head groups
  int tile = blockIdx.x;
  int m0 = (tile / NG) * 16;                // row tile over B*T
  int g  = tile % NG;
  int lane = threadIdx.x & 31;
  int n = lane & 15, hf = lane >> 4;

  const _Float16* wmat;                     // this head's 64 weight rows
  int type;                                 // 0=q 1=k 2=v
  if (g < NHEAD)            { wmat = wqh + (size_t)g * HD * CC;                type = 0; }
  else if (g < NHEAD + NKV) { wmat = wkh + (size_t)(g - NHEAD) * HD * CC;       type = 1; }
  else                      { wmat = wvh + (size_t)(g - NHEAD - NKV) * HD * CC; type = 2; }

  const _Float16* xrow = xh + (size_t)(m0 + n) * CC;

  v8f acc[4] = {};
  for (int kk = 0; kk < CC; kk += 32) {
    int o0 = kk + hf * 8;
    v16h a = frag_f16(xrow + o0, xrow + o0 + 16);
#pragma unroll
    for (int c = 0; c < 4; ++c) {
      const _Float16* wrow = wmat + (size_t)(c * 16 + n) * CC + o0;
      v16h b = frag_f16(wrow, wrow + 16);
      acc[c] = wmma_f16(a, b, acc[c]);
    }
  }

  int b_ = m0 >> 11;                        // batch (tiles never straddle)
  int t0 = m0 & (TT - 1);

  if (type < 2) {                           // RoPE on q and k only
#pragma unroll
    for (int c = 0; c < 4; ++c) {
      int d = c * 16 + n;
      float inv = __expf(-(float)(d & ~1) * (__logf(THETA) / (float)HD));
#pragma unroll
      for (int i = 0; i < 8; ++i) {
        int t = t0 + i + 8 * hf;
        float ang = (float)t * inv;
        float cs = __cosf(ang), sn = __sinf(ang);
        float mine  = acc[c][i];
        float other = __shfl_xor(mine, 1, 32);
        acc[c][i] = (d & 1) ? (other * sn + mine * cs)
                            : (mine * cs - other * sn);
      }
    }
    // transpose through LDS -> packed 128-bit row stores
    _Float16* base = (type == 0)
        ? qb   + ((size_t)(b_ * NHEAD + g) * TT + t0) * HD
        : kbuf + ((size_t)(b_ * NKV + (g - NHEAD)) * TT + t0) * HD;
#pragma unroll
    for (int c = 0; c < 4; ++c)
#pragma unroll
      for (int i = 0; i < 8; ++i)
        tls[(i + 8 * hf) * 72 + c * 16 + n] = (_Float16)acc[c][i];
    __syncthreads();
#pragma unroll
    for (int j = 0; j < 4; ++j) {
      v8h pk = *(const v8h*)&tls[n * 72 + hf * 32 + j * 8];
      *(v8h*)(base + (size_t)n * HD + hf * 32 + j * 8) = pk;
    }
  } else {                                  // v: transposed, direct v8h store
    int kv = g - NHEAD - NKV;
#pragma unroll
    for (int c = 0; c < 4; ++c) {
      int d = c * 16 + n;
      v8h pk;
#pragma unroll
      for (int i = 0; i < 8; ++i) pk[i] = (_Float16)acc[c][i];
      *(v8h*)(vtb + ((size_t)(b_ * NKV + kv) * HD + d) * TT + t0 + 8 * hf) = pk;
    }
  }
}

// ---------------------------------------------------------------------------
// Kernel 2: causal flash attention. One wave per (b, head, 16-row q tile).
// S staged f32 through LDS (C-layout -> A-layout); per-lane half-row softmax
// with one shfl_xor(16) per stat; alpha broadcast via v_readlane.
// ---------------------------------------------------------------------------
__global__ __launch_bounds__(32) void attn_kernel(
    const _Float16* __restrict__ qb, const _Float16* __restrict__ kbuf,
    const _Float16* __restrict__ vtb, _Float16* __restrict__ yb)
{
  __shared__ __align__(16) char smem[16 * 36 * 4];  // 2304 B, dual-purpose
  float*    sls = (float*)smem;                     // 16 rows, stride 36 f32
  _Float16* hls = (_Float16*)smem;                  // 16 rows, stride 72 f16

  int bid = blockIdx.x;
  int qt = bid & 127;                       // T/16 tiles
  int bh = bid >> 7;                        // b*9 + h
  int b = bh / NHEAD, h = bh % NHEAD;
  int kvh = h / (NHEAD / NKV);
  int lane = threadIdx.x & 31;
  int n = lane & 15, hf = lane >> 4;
  int q0 = qt * 16;

  const _Float16* qp = qb   + ((size_t)(b * NHEAD + h)  * TT + q0) * HD;
  const _Float16* kp = kbuf + ((size_t)(b * NKV + kvh)  * TT) * HD;
  const _Float16* vp = vtb  + ((size_t)(b * NKV + kvh)  * HD) * TT;

  v16h qa[2];
#pragma unroll
  for (int c = 0; c < 2; ++c) {
    const _Float16* r = qp + (size_t)n * HD + c * 32 + hf * 8;
    qa[c] = frag_f16(r, r + 16);
  }

  v8f o[4] = {};
  float rmax = -__builtin_inff();           // stats for row n (this lane)
  float rsum = 0.0f;

  const float scale = 0.125f;               // 1/sqrt(64)
  int q_hi = q0 + 15;

  for (int kb0 = 0; kb0 <= q_hi; kb0 += 32) {
    // --- S = Q K^T, 2 x 16-key subtiles, masked+scaled into LDS ----------
#pragma unroll
    for (int sub = 0; sub < 2; ++sub) {
      int kbase = kb0 + sub * 16;
      v8f sc = {};
      if (kbase <= q_hi) {                  // uniform branch
#pragma unroll
        for (int c = 0; c < 2; ++c) {
          const _Float16* r = kp + (size_t)(kbase + n) * HD + c * 32 + hf * 8;
          v16h kf = frag_f16(r, r + 16);
          sc = wmma_f16(qa[c], kf, sc);
        }
      }
#pragma unroll
      for (int i = 0; i < 8; ++i) {
        int row = q0 + i + 8 * hf;
        int col = kbase + n;
        sls[(i + 8 * hf) * 36 + sub * 16 + n] =
            (col <= row) ? sc[i] * scale : -__builtin_inff();
      }
    }
    __syncthreads();

    // --- row n in A-layout: keys hf*8+{0..7} and 16+hf*8+{0..7} ----------
    const float* rb = &sls[n * 36 + hf * 8];
    v4f c0 = *(const v4f*)(rb);
    v4f c1 = *(const v4f*)(rb + 4);
    v4f c2 = *(const v4f*)(rb + 16);
    v4f c3 = *(const v4f*)(rb + 20);
    __syncthreads();

    float lm = -__builtin_inff();
#pragma unroll
    for (int j = 0; j < 4; ++j)
      lm = fmaxf(fmaxf(lm, fmaxf(c0[j], c1[j])), fmaxf(c2[j], c3[j]));
    float mrow = fmaxf(lm, __shfl_xor(lm, 16, 32));
    float mnew = fmaxf(rmax, mrow);
    float alpha = __expf(rmax - mnew);
    rmax = mnew;

    v16h pa;
    float ls = 0.0f;
#pragma unroll
    for (int j = 0; j < 4; ++j) {
      float p0 = __expf(c0[j] - mnew);
      float p1 = __expf(c1[j] - mnew);
      float p2 = __expf(c2[j] - mnew);
      float p3 = __expf(c3[j] - mnew);
      ls += (p0 + p1) + (p2 + p3);
      pa[j]      = (_Float16)p0;
      pa[4 + j]  = (_Float16)p1;
      pa[8 + j]  = (_Float16)p2;
      pa[12 + j] = (_Float16)p3;
    }
    rsum = rsum * alpha + ls + __shfl_xor(ls, 16, 32);

    // --- rescale O rows: per-row alpha via v_readlane broadcasts ---------
    float a_lo[8], a_hi[8];
#pragma unroll
    for (int i = 0; i < 8; ++i) {
      a_lo[i] = readlane_f(alpha, i);
      a_hi[i] = readlane_f(alpha, i + 8);
    }
#pragma unroll
    for (int i = 0; i < 8; ++i) {
      float ar = hf ? a_hi[i] : a_lo[i];
#pragma unroll
      for (int c2i = 0; c2i < 4; ++c2i) o[c2i][i] *= ar;
    }

    // --- O += P * V ------------------------------------------------------
#pragma unroll
    for (int c2i = 0; c2i < 4; ++c2i) {
      const _Float16* vr = vp + (size_t)(c2i * 16 + n) * TT + kb0 + hf * 8;
      v16h vb = frag_f16(vr, vr + 16);
      o[c2i] = wmma_f16(pa, vb, o[c2i]);
    }
  }

  // final 1/rowsum, LDS-transpose, packed 128-bit y stores
  float inv = 1.0f / rsum;
  float i_lo[8], i_hi[8];
#pragma unroll
  for (int i = 0; i < 8; ++i) {
    i_lo[i] = readlane_f(inv, i);
    i_hi[i] = readlane_f(inv, i + 8);
  }
#pragma unroll
  for (int i = 0; i < 8; ++i) {
    float ir = hf ? i_hi[i] : i_lo[i];
#pragma unroll
    for (int c2i = 0; c2i < 4; ++c2i)
      hls[(i + 8 * hf) * 72 + c2i * 16 + n] = (_Float16)(o[c2i][i] * ir);
  }
  __syncthreads();
  _Float16* ybase = yb + ((size_t)b * TT + q0 + n) * (NHEAD * HD) + h * HD;
#pragma unroll
  for (int j = 0; j < 4; ++j) {
    v8h pk = *(const v8h*)&hls[n * 72 + hf * 32 + j * 8];
    *(v8h*)(ybase + hf * 32 + j * 8) = pk;
  }
}

// ---------------------------------------------------------------------------
// Kernel 3: output projection y(8192x576,f16) @ wo^T(576x576,f16) -> f32.
// One wave per (16-row tile, 64-feature group); LDS transpose + b128 stores.
// ---------------------------------------------------------------------------
__global__ __launch_bounds__(32) void outproj_kernel(
    const _Float16* __restrict__ yb, const _Float16* __restrict__ woh,
    float* __restrict__ out)
{
  __shared__ __align__(16) float ols[16 * 68];      // stride 68 f32

  const int NG = CC / 64;                   // 9 feature groups
  int tile = blockIdx.x;
  int m0 = (tile / NG) * 16;
  int n0 = (tile % NG) * 64;
  int lane = threadIdx.x & 31;
  int n = lane & 15, hf = lane >> 4;

  const _Float16* yrow = yb + (size_t)(m0 + n) * CC;

  v8f acc[4] = {};
  for (int kk = 0; kk < CC; kk += 32) {
    int o0 = kk + hf * 8;
    v16h a = frag_f16(yrow + o0, yrow + o0 + 16);
#pragma unroll
    for (int c = 0; c < 4; ++c) {
      const _Float16* wrow = woh + (size_t)(n0 + c * 16 + n) * CC + o0;
      v16h b = frag_f16(wrow, wrow + 16);
      acc[c] = wmma_f16(a, b, acc[c]);
    }
  }
#pragma unroll
  for (int c = 0; c < 4; ++c)
#pragma unroll
    for (int i = 0; i < 8; ++i)
      ols[(i + 8 * hf) * 68 + c * 16 + n] = acc[c][i];
  __syncthreads();
  float* obase = out + (size_t)(m0 + n) * CC + n0 + hf * 32;
#pragma unroll
  for (int j = 0; j < 8; ++j)
    *(v4f*)(obase + j * 4) = *(const v4f*)&ols[n * 68 + hf * 32 + j * 4];
}

// ---------------------------------------------------------------------------
extern "C" void kernel_launch(void* const* d_in, const int* in_sizes, int n_in,
                              void* d_out, int out_size, void* d_ws, size_t ws_size,
                              hipStream_t stream) {
  const float* x  = (const float*)d_in[0];
  const float* wq = (const float*)d_in[1];
  const float* wk = (const float*)d_in[2];
  const float* wv = (const float*)d_in[3];
  const float* wo = (const float*)d_in[4];
  float* out = (float*)d_out;

  const size_t xN  = (size_t)BB * TT * CC;          // 4718592
  const size_t wqN = (size_t)NHEAD * HD * CC;       // 331776
  const size_t wkN = (size_t)NKV * HD * CC;         // 110592
  const size_t woN = (size_t)CC * CC;               // 331776

  _Float16* qb  = (_Float16*)d_ws;
  _Float16* kb  = qb  + (size_t)BB * NHEAD * TT * HD;
  _Float16* vtb = kb  + (size_t)BB * NKV   * TT * HD;
  _Float16* yb  = vtb + (size_t)BB * NKV   * TT * HD;
  _Float16* xh  = yb  + xN;                         // y: [B*T, 576] f16
  _Float16* wqh = xh  + xN;
  _Float16* wkh = wqh + wqN;
  _Float16* wvh = wkh + wkN;
  _Float16* woh = wvh + wkN;

  dim3 blk(32);
  // precision-convert inputs once
  cvt_f32_to_f16<<<dim3((xN / 8 + 255) / 256), dim3(256), 0, stream>>>(x, xh, (int)(xN / 8));
  cvt_f32_to_f16<<<dim3((wqN / 8 + 255) / 256), dim3(256), 0, stream>>>(wq, wqh, (int)(wqN / 8));
  cvt_f32_to_f16<<<dim3((wkN / 8 + 255) / 256), dim3(256), 0, stream>>>(wk, wkh, (int)(wkN / 8));
  cvt_f32_to_f16<<<dim3((wkN / 8 + 255) / 256), dim3(256), 0, stream>>>(wv, wvh, (int)(wkN / 8));
  cvt_f32_to_f16<<<dim3((woN / 8 + 255) / 256), dim3(256), 0, stream>>>(wo, woh, (int)(woN / 8));

  qkv_rope_kernel<<<dim3((BB * TT / 16) * (NHEAD + 2 * NKV)), blk, 0, stream>>>(
      xh, wqh, wkh, wvh, qb, kb, vtb);
  attn_kernel<<<dim3(BB * NHEAD * (TT / 16)), blk, 0, stream>>>(qb, kb, vtb, yb);
  outproj_kernel<<<dim3((BB * TT / 16) * (CC / 64)), blk, 0, stream>>>(yb, woh, out);
}
